// tempMP5_52329881534846
// MI455X (gfx1250) — matrix operand order repaired and checked by reading, not compiled, not run on hardware
//
#include <hip/hip_runtime.h>
#include <math.h>

#define NT 2000
#define NW 50000
#define ND 20000
#define DD 256
#define HH 8
#define DKK 32
#define E_TT 128000
#define E_WT 256000
#define E_TD 320000
#define E_WD 320000

typedef __attribute__((ext_vector_type(16))) _Float16 v16h;
typedef __attribute__((ext_vector_type(8)))  _Float16 v8h;
typedef __attribute__((ext_vector_type(8)))  float    v8f;

// ---------------------------------------------------------------------------
// GEMM:  C[N x 256] = Ah[N x 256](f16) @ P(packed f16 W) + bias
// One wave per 16x16 tile; inner loop = 4x global_load_b128 + 1 wmma.
// Packed layout: P[((ntile*8 + kt)*32 + lane)*16 + e] =
//                (f16) W[(kt*32 + e + 16*(lane>>4)) * 256 + ntile*16 + (lane&15)]
// ---------------------------------------------------------------------------
__global__ void __launch_bounds__(256) gemm256_wmma(
    const _Float16* __restrict__ Ah, const _Float16* __restrict__ P,
    const float* __restrict__ bias, float* __restrict__ C, int N)
{
    const int wave  = threadIdx.x >> 5;
    const int lane  = threadIdx.x & 31;
    const int half  = lane >> 4;
    const int row   = lane & 15;
    const int m0    = blockIdx.x * 16;
    const int ntile = blockIdx.y * 8 + wave;

    int arow = m0 + row;
    if (arow >= N) arow = N - 1;               // clamp loads; stores guarded
    const _Float16* Arow = Ah + (size_t)arow * DD;
    const _Float16* Pw   = P + (size_t)ntile * 4096 + lane * 16;

    v8f c = {};
    #pragma unroll
    for (int kt = 0; kt < 8; ++kt) {
        // A fragment: halves [K0+16h .. K0+16h+15] (contiguous)
        const v8h alo = *(const v8h*)(Arow + kt * 32 + 16 * half);
        const v8h ahi = *(const v8h*)(Arow + kt * 32 + 16 * half + 8);
        // B fragment: 16 contiguous packed halves
        const v8h blo = *(const v8h*)(Pw + kt * 512);
        const v8h bhi = *(const v8h*)(Pw + kt * 512 + 8);
        const v16h a = __builtin_shufflevector(alo, ahi,
                0,1,2,3,4,5,6,7,8,9,10,11,12,13,14,15);
        const v16h b = __builtin_shufflevector(blo, bhi,
                0,1,2,3,4,5,6,7,8,9,10,11,12,13,14,15);
        c = __builtin_amdgcn_wmma_f32_16x16x32_f16(
                false, a, false, b, (short)0, c, false, false);
    }
    const int col = ntile * 16 + row;
    const float bv = bias[col];
    #pragma unroll
    for (int i = 0; i < 8; ++i) {
        const int r = m0 + i + 8 * half;
        if (r < N) C[(size_t)r * DD + col] = c[i] + bv;
    }
}

// Pack fp32 W[256x256] (row-major, K x N) into f16 B-fragment order.
__global__ void pack_w(const float* __restrict__ W, _Float16* __restrict__ P)
{
    const int idx = blockIdx.x * blockDim.x + threadIdx.x;
    if (idx >= DD * DD) return;
    const int e     = idx & 15;
    const int lane  = (idx >> 4) & 31;
    const int kt    = (idx >> 9) & 7;
    const int ntile = idx >> 12;
    const int kk    = kt * 32 + e + 16 * (lane >> 4);
    P[idx] = (_Float16)W[(size_t)kk * DD + ntile * 16 + (lane & 15)];
}

// fp32 -> f16 row-major conversion
__global__ void conv_h(const float* __restrict__ in, _Float16* __restrict__ out, int n)
{
    const int i = blockIdx.x * blockDim.x + threadIdx.x;
    if (i < n) out[i] = (_Float16)in[i];
}

// ---------------------------------------------------------------------------
// Fold per-head relation matrix into projection (fp32, tiny, runs once each):
//   Wc[k][h*32+o] = sum_d Wp[k][h*32+d] * Amat[h][d][o]   (+ same for bias)
// ---------------------------------------------------------------------------
__global__ void combine_w(const float* __restrict__ Wp, const float* __restrict__ bp,
                          const float* __restrict__ Amat,
                          float* __restrict__ Wc, float* __restrict__ bc)
{
    const int idx = blockIdx.x * blockDim.x + threadIdx.x;
    if (idx < DD * DD) {
        const int k = idx >> 8, col = idx & 255, h = col >> 5, o = col & 31;
        float acc = 0.f;
        #pragma unroll
        for (int d = 0; d < DKK; ++d)
            acc += Wp[k * DD + h * DKK + d] * Amat[h * 1024 + d * DKK + o];
        Wc[idx] = acc;
    } else if (idx < DD * DD + DD) {
        const int col = idx - DD * DD, h = col >> 5, o = col & 31;
        float acc = 0.f;
        #pragma unroll
        for (int d = 0; d < DKK; ++d)
            acc += bp[h * DKK + d] * Amat[h * 1024 + d * DKK + o];
        bc[col] = acc;
    }
}

// ---------------------------------------------------------------------------
// Edge softmax pipeline (thread per edge-head pair), all L2-resident
// ---------------------------------------------------------------------------
__device__ __forceinline__ void atomicMaxF(float* addr, float val)
{
    if (val >= 0.f) atomicMax((int*)addr, __float_as_int(val));
    else            atomicMin((unsigned int*)addr, __float_as_uint(val));
}

__global__ void fillk(float* __restrict__ p, float v, int n)
{
    const int i = blockIdx.x * blockDim.x + threadIdx.x;
    if (i < n) p[i] = v;
}

__global__ void edge_att(const int* __restrict__ src, const int* __restrict__ dst,
                         const float* __restrict__ key, const float* __restrict__ q,
                         const float* __restrict__ pri, float scale,
                         float* __restrict__ attb, float* __restrict__ m, int E)
{
    const int idx = blockIdx.x * blockDim.x + threadIdx.x;
    if (idx >= E * HH) return;
    const int e = idx >> 3, h = idx & 7;
    const int s = src[e], d = dst[e];
    const float4* qp = (const float4*)(q   + (size_t)d * DD + h * DKK);
    const float4* kp = (const float4*)(key + (size_t)s * DD + h * DKK);
    float acc = 0.f;
    #pragma unroll
    for (int j = 0; j < 8; ++j) {
        const float4 a = qp[j], b = kp[j];
        acc += a.x * b.x + a.y * b.y + a.z * b.z + a.w * b.w;
    }
    const float att = acc * pri[h] * scale;
    attb[idx] = att;
    atomicMaxF(&m[d * HH + h], att);
}

__global__ void edge_expsum(const int* __restrict__ dst, float* __restrict__ attb,
                            const float* __restrict__ m, float* __restrict__ ssum, int E)
{
    const int idx = blockIdx.x * blockDim.x + threadIdx.x;
    if (idx >= E * HH) return;
    const int e = idx >> 3, h = idx & 7;
    const int d = dst[e];
    const float ev = __expf(attb[idx] - m[d * HH + h]);
    attb[idx] = ev;
    atomicAdd(&ssum[d * HH + h], ev);
}

__global__ void edge_scatter(const int* __restrict__ src, const int* __restrict__ dst,
                             const float* __restrict__ attb, const float* __restrict__ ssum,
                             const float* __restrict__ val, float* __restrict__ t, int E)
{
    const int idx = blockIdx.x * blockDim.x + threadIdx.x;
    if (idx >= E * HH) return;
    const int e = idx >> 3, h = idx & 7;
    const int s = src[e], d = dst[e];
    const float a = attb[idx] / ssum[d * HH + h];
    const float4* vp = (const float4*)(val + (size_t)s * DD + h * DKK);
    float* tp = t + (size_t)d * DD + h * DKK;
    #pragma unroll
    for (int j = 0; j < 8; ++j) {
        const float4 v = vp[j];
        atomicAdd(tp + j * 4 + 0, a * v.x);
        atomicAdd(tp + j * 4 + 1, a * v.y);
        atomicAdd(tp + j * 4 + 2, a * v.z);
        atomicAdd(tp + j * 4 + 3, a * v.w);
    }
}

// ---------------------------------------------------------------------------
// Elementwise epilogues (emit f16 where the consumer is a GEMM)
// ---------------------------------------------------------------------------
__global__ void relu_mean_h(const float* __restrict__ a, const float* __restrict__ b,
                            _Float16* __restrict__ oh, int n)
{
    const int i = blockIdx.x * blockDim.x + threadIdx.x;
    if (i < n) oh[i] = (_Float16)(0.5f * (fmaxf(a[i], 0.f) + fmaxf(b[i], 0.f)));
}

__global__ void mix_k_h(const float* __restrict__ trans, const float* __restrict__ feat,
                        const float* __restrict__ skip, int sidx,
                        _Float16* __restrict__ oh, int n)
{
    const int i = blockIdx.x * blockDim.x + threadIdx.x;
    if (i < n) {
        const float al = 1.f / (1.f + __expf(-skip[sidx]));
        oh[i] = (_Float16)(trans[i] * al + feat[i] * (1.f - al));
    }
}

__global__ void mix_k(const float* __restrict__ trans, const float* __restrict__ feat,
                      const float* __restrict__ skip, int sidx,
                      float* __restrict__ o, int n)
{
    const int i = blockIdx.x * blockDim.x + threadIdx.x;
    if (i < n) {
        const float al = 1.f / (1.f + __expf(-skip[sidx]));
        o[i] = trans[i] * al + feat[i] * (1.f - al);
    }
}

// x = (tanh?)(A + B); out = LayerNorm(x)*g + b; optional f16 mirror for GEMMs
__global__ void __launch_bounds__(256) ln_k(const float* __restrict__ A,
                                            const float* __restrict__ B, int doTanh,
                                            const float* __restrict__ g,
                                            const float* __restrict__ b,
                                            float* __restrict__ out,
                                            _Float16* __restrict__ outh)
{
    const int rowBase = blockIdx.x * DD;
    const int t = threadIdx.x;
    float x = A[rowBase + t];
    if (B) x += B[rowBase + t];
    if (doTanh) x = tanhf(x);

    __shared__ float red[DD];
    red[t] = x;
    __syncthreads();
    #pragma unroll
    for (int s = 128; s > 0; s >>= 1) { if (t < s) red[t] += red[t + s]; __syncthreads(); }
    const float mu = red[0] * (1.f / DD);
    __syncthreads();
    const float dx = x - mu;
    red[t] = dx * dx;
    __syncthreads();
    #pragma unroll
    for (int s = 128; s > 0; s >>= 1) { if (t < s) red[t] += red[t + s]; __syncthreads(); }
    const float var = red[0] * (1.f / DD);
    const float y = dx * rsqrtf(var + 1e-5f) * g[t] + b[t];
    out[rowBase + t] = y;
    if (outh) outh[rowBase + t] = (_Float16)y;
}

// ---------------------------------------------------------------------------
extern "C" void kernel_launch(void* const* d_in, const int* in_sizes, int n_in,
                              void* d_out, int out_size, void* d_ws, size_t ws_size,
                              hipStream_t stream)
{
    (void)in_sizes; (void)n_in; (void)out_size; (void)ws_size;

    const float* feat_topic = (const float*)d_in[0];
    const float* feat_word  = (const float*)d_in[1];
    const float* feat_doc   = (const float*)d_in[2];
    const float* ht_prev    = (const float*)d_in[3];
    const float* Wk   = (const float*)d_in[4];
    const float* bk   = (const float*)d_in[5];
    const float* Wq   = (const float*)d_in[6];
    const float* bq   = (const float*)d_in[7];
    const float* Wv   = (const float*)d_in[8];
    const float* bv   = (const float*)d_in[9];
    const float* Wa   = (const float*)d_in[10];
    const float* ba   = (const float*)d_in[11];
    const float* ln_g = (const float*)d_in[12];
    const float* ln_b = (const float*)d_in[13];
    const float* skip = (const float*)d_in[14];
    const float* Wih  = (const float*)d_in[15];
    const float* Whh  = (const float*)d_in[16];
    const float* bih  = (const float*)d_in[17];
    const float* bhh  = (const float*)d_in[18];
    const float* rel_pri = (const float*)d_in[19];
    const float* rel_att = (const float*)d_in[20];
    const float* rel_msg = (const float*)d_in[21];
    const int* src_tt = (const int*)d_in[22];
    const int* dst_tt = (const int*)d_in[23];
    const int* src_wt = (const int*)d_in[24];
    const int* dst_wt = (const int*)d_in[25];
    const int* src_td = (const int*)d_in[26];
    const int* dst_td = (const int*)d_in[27];
    const int* src_wd = (const int*)d_in[28];
    const int* dst_wd = (const int*)d_in[29];

    float* topic_out = (float*)d_out;
    float* doc_out   = (float*)d_out + (size_t)NT * DD;

    // ---- scratch layout (float units; all sizes multiples of 256 -> aligned) ----
    float* ws = (float*)d_ws;
    size_t off = 0;
    auto alloc  = [&](size_t n) { float* p = ws + off; off += n; return p; };
    auto allocH = [&](size_t nh) { _Float16* p = (_Float16*)(ws + off); off += nh / 2; return p; };

    float* q_t    = alloc((size_t)NT * DD);
    float* q_d    = alloc((size_t)ND * DD);
    float* key_tt = alloc((size_t)NT * DD);
    float* val_tt = alloc((size_t)NT * DD);
    float* key_wt = alloc((size_t)NW * DD);
    float* val_wt = alloc((size_t)NW * DD);
    float* key_wd = alloc((size_t)NW * DD);
    float* val_wd = alloc((size_t)NW * DD);
    float* key_td = alloc((size_t)NT * DD);
    float* val_td = alloc((size_t)NT * DD);
    float* t_tt   = alloc((size_t)NT * DD);
    float* t_wt   = alloc((size_t)NT * DD);
    float* t_td   = alloc((size_t)ND * DD);
    float* t_wd   = alloc((size_t)ND * DD);
    float* trans  = alloc((size_t)NT * DD);
    float* g1     = alloc((size_t)NT * DD);
    float* g2     = alloc((size_t)NT * DD);
    float* trans2 = alloc((size_t)ND * DD);
    float* attb   = alloc((size_t)E_TD * HH);
    float* mbuf   = alloc((size_t)ND * HH);
    float* sbuf   = alloc((size_t)ND * HH);
    float* Wc[8]; float* bc[8];
    for (int i = 0; i < 8; ++i) { Wc[i] = alloc(DD * DD); bc[i] = alloc(DD); }

    _Float16* fh_topic    = allocH((size_t)NT * DD);
    _Float16* fh_word     = allocH((size_t)NW * DD);
    _Float16* fh_doc      = allocH((size_t)ND * DD);
    _Float16* fh_ht       = allocH((size_t)NT * DD);
    _Float16* fh_ttopic   = allocH((size_t)NT * DD);
    _Float16* fh_mix      = allocH((size_t)NT * DD);
    _Float16* fh_topicout = allocH((size_t)NT * DD);
    _Float16* fh_tdoc     = allocH((size_t)ND * DD);
    _Float16* P[14];
    for (int i = 0; i < 14; ++i) P[i] = allocH(DD * DD);
    // P: 0=Wq0 1=Wq2 2..9=Wc0..7 10=Wa0 11=Wa1 12=Wih 13=Whh

    const float SCALE = 0.17677669529663687f; // 1/sqrt(32)

    auto gemm = [&](const _Float16* Ah, const _Float16* Pk, const float* bias,
                    float* C, int N) {
        dim3 g((N + 15) / 16, 2), b(256);
        gemm256_wmma<<<g, b, 0, stream>>>(Ah, Pk, bias, C, N);
    };
    auto pack = [&](const float* W, _Float16* Pk) {
        pack_w<<<(DD * DD + 255) / 256, 256, 0, stream>>>(W, Pk);
    };
    auto relation = [&](const int* src, const int* dst, int E, const float* key,
                        const float* val, const float* q, const float* pri,
                        float* t, int n_dst) {
        const int nEH = E * HH, nMS = n_dst * HH, nT = n_dst * DD;
        fillk<<<(nMS + 255) / 256, 256, 0, stream>>>(mbuf, -INFINITY, nMS);
        fillk<<<(nMS + 255) / 256, 256, 0, stream>>>(sbuf, 0.f, nMS);
        fillk<<<(nT + 255) / 256, 256, 0, stream>>>(t, 0.f, nT);
        edge_att<<<(nEH + 255) / 256, 256, 0, stream>>>(src, dst, key, q, pri, SCALE, attb, mbuf, E);
        edge_expsum<<<(nEH + 255) / 256, 256, 0, stream>>>(dst, attb, mbuf, sbuf, E);
        edge_scatter<<<(nEH + 255) / 256, 256, 0, stream>>>(src, dst, attb, sbuf, val, t, E);
    };

    // ---- activation conversions (feat_word converted once, feeds 4 GEMMs) ----
    conv_h<<<(NT * DD + 255) / 256, 256, 0, stream>>>(feat_topic, fh_topic, NT * DD);
    conv_h<<<(NW * DD + 255) / 256, 256, 0, stream>>>(feat_word,  fh_word,  NW * DD);
    conv_h<<<(ND * DD + 255) / 256, 256, 0, stream>>>(feat_doc,   fh_doc,   ND * DD);
    conv_h<<<(NT * DD + 255) / 256, 256, 0, stream>>>(ht_prev,    fh_ht,    NT * DD);

    // ---- fold (W_proj, A_rel) into combined weights, then pack all weights ----
    const int cb = (DD * DD + DD + 255) / 256;
    combine_w<<<cb, 256, 0, stream>>>(Wk,         bk,       rel_att,            Wc[0], bc[0]);
    combine_w<<<cb, 256, 0, stream>>>(Wv,         bv,       rel_msg,            Wc[1], bc[1]);
    combine_w<<<cb, 256, 0, stream>>>(Wk + 65536, bk + 256, rel_att + 8192,     Wc[2], bc[2]);
    combine_w<<<cb, 256, 0, stream>>>(Wv + 65536, bv + 256, rel_msg + 8192,     Wc[3], bc[3]);
    combine_w<<<cb, 256, 0, stream>>>(Wk,         bk,       rel_att + 2 * 8192, Wc[4], bc[4]);
    combine_w<<<cb, 256, 0, stream>>>(Wv,         bv,       rel_msg + 2 * 8192, Wc[5], bc[5]);
    combine_w<<<cb, 256, 0, stream>>>(Wk + 65536, bk + 256, rel_att + 3 * 8192, Wc[6], bc[6]);
    combine_w<<<cb, 256, 0, stream>>>(Wv + 65536, bv + 256, rel_msg + 3 * 8192, Wc[7], bc[7]);

    pack(Wq,             P[0]);
    pack(Wq + 2 * 65536, P[1]);
    for (int i = 0; i < 8; ++i) pack(Wc[i], P[2 + i]);
    pack(Wa,         P[10]);
    pack(Wa + 65536, P[11]);
    pack(Wih,        P[12]);
    pack(Whh,        P[13]);

    // ---- phase 1 projections (WMMA GEMMs) ----
    gemm(fh_topic, P[0], bq,        q_t,    NT);
    gemm(fh_doc,   P[1], bq + 512,  q_d,    ND);
    gemm(fh_topic, P[2], bc[0],     key_tt, NT);
    gemm(fh_topic, P[3], bc[1],     val_tt, NT);
    gemm(fh_word,  P[4], bc[2],     key_wt, NW);
    gemm(fh_word,  P[5], bc[3],     val_wt, NW);
    gemm(fh_word,  P[8], bc[6],     key_wd, NW);
    gemm(fh_word,  P[9], bc[7],     val_wd, NW);

    // ---- phase 1 edge softmax + aggregate into topics ----
    relation(src_tt, dst_tt, E_TT, key_tt, val_tt, q_t, rel_pri,      t_tt, NT);
    relation(src_wt, dst_wt, E_WT, key_wt, val_wt, q_t, rel_pri + HH, t_wt, NT);

    // ---- topic update: relu-mean -> a_linear -> skip mix -> RNNCell -> LN ----
    const int nTopic = NT * DD;
    relu_mean_h<<<(nTopic + 255) / 256, 256, 0, stream>>>(t_tt, t_wt, fh_ttopic, nTopic);
    gemm(fh_ttopic, P[10], ba, trans, NT);
    mix_k_h<<<(nTopic + 255) / 256, 256, 0, stream>>>(trans, feat_topic, skip, 0, fh_mix, nTopic);
    gemm(fh_mix, P[12], bih, g1, NT);
    gemm(fh_ht,  P[13], bhh, g2, NT);
    ln_k<<<NT, 256, 0, stream>>>(g1, g2, 1, ln_g, ln_b, topic_out, fh_topicout);

    // ---- phase 2: topic->doc and word->doc messages ----
    gemm(fh_topicout, P[6], bc[4], key_td, NT);
    gemm(fh_topicout, P[7], bc[5], val_td, NT);
    relation(src_td, dst_td, E_TD, key_td, val_td, q_d, rel_pri + 2 * HH, t_td, ND);
    relation(src_wd, dst_wd, E_WD, key_wd, val_wd, q_d, rel_pri + 3 * HH, t_wd, ND);

    // ---- doc update: relu-mean -> a_linear -> skip mix -> LN ----
    const int nDoc = ND * DD;
    relu_mean_h<<<(nDoc + 255) / 256, 256, 0, stream>>>(t_td, t_wd, fh_tdoc, nDoc);
    gemm(fh_tdoc, P[11], ba + 256, trans2, ND);
    mix_k<<<(nDoc + 255) / 256, 256, 0, stream>>>(trans2, feat_doc, skip, 1, trans2, nDoc);
    ln_k<<<ND, 256, 0, stream>>>(trans2, nullptr, 0, ln_g + 256, ln_b + 256, doc_out, nullptr);
}
